// FixedMillennialAiReasoningEngine_20993800143199
// MI455X (gfx1250) — matrix operand order repaired and checked by reading, not compiled, not run on hardware
//
#include <hip/hip_runtime.h>
#include <hip/hip_bf16.h>
#include <stdint.h>

// Problem constants (fixed by the reference): B=1, S=2048, H=1024, M=1024, D=12
#define S_LEN 2048
#define HDIM  1024
#define DDEPTH 12

typedef __attribute__((ext_vector_type(16))) __bf16       v16bf;
typedef __attribute__((ext_vector_type(8)))  float        v8f;
typedef __attribute__((ext_vector_type(4)))  uint32_t     v4u;
typedef __attribute__((ext_vector_type(4)))  unsigned int v4ui;
typedef __attribute__((ext_vector_type(8)))  int          v8i;
typedef __attribute__((ext_vector_type(4)))  int          v4i;

// ---- Tensor Data Mover availability (device pass only; host pass falls back) ----
#ifdef __has_builtin
#  if __has_builtin(__builtin_amdgcn_tensor_load_to_lds) && __has_builtin(__builtin_amdgcn_s_wait_tensorcnt)
#    define USE_TDM 1
#  else
#    define USE_TDM 0
#  endif
#else
#  define USE_TDM 0
#endif

#if USE_TDM
// 2-D tile load Global->LDS via TDM (D# groups per CDNA5 ISA ch.8).
// data_size = 2 bytes (bf16). tile_w elems contiguous (dim0), tile_h rows (dim1),
// row stride = stride_elems. lds_off = byte offset into workgroup LDS.
__device__ __forceinline__ void tdm_load_tile_2d(unsigned int lds_off,
                                                 unsigned long long gaddr,
                                                 unsigned int tile_w, unsigned int tile_h,
                                                 unsigned int tensor_w, unsigned int tensor_h,
                                                 unsigned int stride_elems) {
  v4ui g0;
  g0[0] = 1u;                                              // count=1, user descriptor
  g0[1] = lds_off;                                         // lds_addr (bytes)
  g0[2] = (unsigned int)(gaddr & 0xFFFFFFFFull);           // global_addr[31:0]
  g0[3] = (unsigned int)((gaddr >> 32) & 0x01FFFFFFull)    // global_addr[56:32]
          | (2u << 30);                                    // type=2 ("image")
  v8i g1;
  g1[0] = (int)(1u << 16);                                 // wg_mask=0 | data_size=1 (2B)
  g1[1] = (int)((tensor_w & 0xFFFFu) << 16);               // abar=0 | tensor_dim0[15:0]
  g1[2] = (int)(((tensor_w >> 16) & 0xFFFFu) | ((tensor_h & 0xFFFFu) << 16));
  g1[3] = (int)(((tensor_h >> 16) & 0xFFFFu) | ((tile_w & 0xFFFFu) << 16));  // tile_dim0
  g1[4] = (int)(tile_h & 0xFFFFu);                         // tile_dim1 | tile_dim2=0
  g1[5] = (int)stride_elems;                               // tensor_dim0_stride[31:0]
  g1[6] = 0;                                               // stride[47:32] | dim1_stride lo
  g1[7] = 0;
  v4i gz = 0;                                              // groups 2/3: unused (2-D)
#if __clang_major__ >= 23
  v8i z8 = 0;
  __builtin_amdgcn_tensor_load_to_lds(g0, g1, gz, gz, z8, 0);
#else
  __builtin_amdgcn_tensor_load_to_lds(g0, g1, gz, gz, 0);
#endif
}
#endif  // USE_TDM

// ---------- helpers ----------
__device__ __forceinline__ unsigned short f32_to_bf16_rne(float f) {
  uint32_t u = __float_as_uint(f);
  uint32_t r = u + 0x7FFFu + ((u >> 16) & 1u);
  return (unsigned short)(r >> 16);
}

__global__ void cvt_f32_bf16_kernel(const float* __restrict__ src,
                                    unsigned short* __restrict__ dst, long n) {
  long i = (long)blockIdx.x * blockDim.x + threadIdx.x;
  long stride = (long)gridDim.x * blockDim.x;
  for (; i < n; i += stride) dst[i] = f32_to_bf16_rne(src[i]);
}

// ---------- WMMA GEMM:  C[M,N] = act( A[M,K] * B[N,K]^T + bias )  (bf16 in, f32 acc) ----------
// Block: 256 threads (8 waves). Block tile 128(M) x 64(N); each wave: 16x64 via 4 accumulators.
// B tile (64 rows x 32 K, 4KB) is staged to LDS by the Tensor Data Mover each K-chunk.
template <bool BIAS, bool RELU, bool OUTBF>
__global__ __launch_bounds__(256)
void gemm_bt_wmma(const unsigned short* __restrict__ A, int lda,
                  const unsigned short* __restrict__ B, int ldb, int nrowsB,
                  const float* __restrict__ bias,
                  void* __restrict__ out, int ldo, int K) {
  const int lane = threadIdx.x & 31;
  const int wave = threadIdx.x >> 5;
  const int r16  = lane & 15;
  const int hi   = lane >> 4;
  const int mBase = blockIdx.y * 128 + wave * 16;
  const int nBase = blockIdx.x * 64;

  const unsigned short* aptr = A + (size_t)(mBase + r16) * lda;

  v8f acc[4] = {v8f{}, v8f{}, v8f{}, v8f{}};
  union Frag { v4u q[2]; v16bf v; };

#if USE_TDM
  __shared__ unsigned short btile[64 * 32];   // [n within tile][k within chunk]
#endif

  for (int kc = 0; kc < K; kc += 32) {
#if USE_TDM
    if (wave == 0) {
      tdm_load_tile_2d(0u,
                       (unsigned long long)(uintptr_t)(B + (size_t)nBase * ldb + kc),
                       /*tile_w=*/32u, /*tile_h=*/64u,
                       /*tensor_w=*/(unsigned)ldb, /*tensor_h=*/(unsigned)nrowsB,
                       /*stride=*/(unsigned)ldb);
      __builtin_amdgcn_s_wait_tensorcnt(0);
    }
    __syncthreads();
#endif
    Frag a;
    a.q[0] = *(const v4u*)(aptr + kc + hi * 16);       // K = kc+hi*16 .. +7
    a.q[1] = *(const v4u*)(aptr + kc + hi * 16 + 8);   // K = kc+hi*16+8 .. +15
    if (kc + 256 < K) __builtin_prefetch(aptr + kc + 256, 0, 0);  // global_prefetch_b8
#pragma unroll
    for (int j = 0; j < 4; ++j) {
      Frag b;
#if USE_TDM
      const unsigned short* bp = &btile[(j * 16 + r16) * 32 + hi * 16];
      b.q[0] = *(const v4u*)(bp);
      b.q[1] = *(const v4u*)(bp + 8);
#else
      const unsigned short* bp = B + (size_t)(nBase + j * 16 + r16) * ldb + kc + hi * 16;
      b.q[0] = *(const v4u*)(bp);
      b.q[1] = *(const v4u*)(bp + 8);
#endif
      acc[j] = __builtin_amdgcn_wmma_f32_16x16x32_bf16(
          false, a.v, false, b.v, (short)0, acc[j], false, false);
    }
#if USE_TDM
    __syncthreads();
#endif
  }

#pragma unroll
  for (int j = 0; j < 4; ++j) {
    const int n = nBase + j * 16 + r16;
    float bn = 0.0f;
    if (BIAS) bn = bias[n];
#pragma unroll
    for (int r = 0; r < 8; ++r) {
      const int m = mBase + hi * 8 + r;
      float v = acc[j][r] + bn;
      if (RELU) v = v > 0.0f ? v : 0.0f;
      if (OUTBF)
        ((unsigned short*)out)[(size_t)m * ldo + n] = f32_to_bf16_rne(v);
      else
        ((float*)out)[(size_t)m * ldo + n] = v;
    }
  }
}

// ---------- WMMA GEMM:  C[M,N] = addsrc + alpha * (A[M,K] * B[K,N])  ----------
// B tile (32 K-rows x 64 N, 4KB) staged to LDS by TDM; column fragments gathered from LDS.
__global__ __launch_bounds__(256)
void gemm_nn_wmma_addscale(const unsigned short* __restrict__ A, int lda,
                           const unsigned short* __restrict__ B, int ldb, int nrowsB,
                           const float* __restrict__ addsrc,
                           float* __restrict__ out, int ldo, int K, float alpha) {
  const int lane = threadIdx.x & 31;
  const int wave = threadIdx.x >> 5;
  const int r16  = lane & 15;
  const int hi   = lane >> 4;
  const int mBase = blockIdx.y * 128 + wave * 16;
  const int nBase = blockIdx.x * 64;

  const unsigned short* aptr = A + (size_t)(mBase + r16) * lda;

  v8f acc[4] = {v8f{}, v8f{}, v8f{}, v8f{}};
  union FragA { v4u q[2]; v16bf v; };
  union FragB { unsigned short s[16]; v16bf v; };

#if USE_TDM
  __shared__ unsigned short btile[32 * 64];   // [k within chunk][n within tile]
#endif

  for (int kc = 0; kc < K; kc += 32) {
#if USE_TDM
    if (wave == 0) {
      tdm_load_tile_2d(0u,
                       (unsigned long long)(uintptr_t)(B + (size_t)kc * ldb + nBase),
                       /*tile_w=*/64u, /*tile_h=*/32u,
                       /*tensor_w=*/(unsigned)ldb, /*tensor_h=*/(unsigned)nrowsB,
                       /*stride=*/(unsigned)ldb);
      __builtin_amdgcn_s_wait_tensorcnt(0);
    }
    __syncthreads();
#endif
    FragA a;
    a.q[0] = *(const v4u*)(aptr + kc + hi * 16);
    a.q[1] = *(const v4u*)(aptr + kc + hi * 16 + 8);
    if (kc + 256 < K) __builtin_prefetch(aptr + kc + 256, 0, 0);
#pragma unroll
    for (int j = 0; j < 4; ++j) {
      FragB b;
#if USE_TDM
#pragma unroll
      for (int i = 0; i < 16; ++i) b.s[i] = btile[(hi * 16 + i) * 64 + j * 16 + r16];
#else
      const int k0 = kc + hi * 16;
      const int col = nBase + j * 16 + r16;
#pragma unroll
      for (int i = 0; i < 16; ++i) b.s[i] = B[(size_t)(k0 + i) * ldb + col];
#endif
      acc[j] = __builtin_amdgcn_wmma_f32_16x16x32_bf16(
          false, a.v, false, b.v, (short)0, acc[j], false, false);
    }
#if USE_TDM
    __syncthreads();
#endif
  }

#pragma unroll
  for (int j = 0; j < 4; ++j) {
    const int n = nBase + j * 16 + r16;
#pragma unroll
    for (int r = 0; r < 8; ++r) {
      const int m = mBase + hi * 8 + r;
      out[(size_t)m * ldo + n] = addsrc[(size_t)m * ldo + n] + alpha * acc[j][r];
    }
  }
}

// ---------- row softmax: f32 scores -> bf16 attn ----------
__global__ __launch_bounds__(256)
void softmax_rows_kernel(const float* __restrict__ scores,
                         unsigned short* __restrict__ attn, int T) {
  const int row = blockIdx.x;
  const float* s = scores + (size_t)row * T;
  unsigned short* o = attn + (size_t)row * T;
  __shared__ float red[256];
  float mx = -3.4e38f;
  for (int i = threadIdx.x; i < T; i += 256) mx = fmaxf(mx, s[i]);
  red[threadIdx.x] = mx; __syncthreads();
  for (int off = 128; off > 0; off >>= 1) {
    if ((int)threadIdx.x < off) red[threadIdx.x] = fmaxf(red[threadIdx.x], red[threadIdx.x + off]);
    __syncthreads();
  }
  mx = red[0]; __syncthreads();
  float sum = 0.0f;
  for (int i = threadIdx.x; i < T; i += 256) sum += __expf(s[i] - mx);
  red[threadIdx.x] = sum; __syncthreads();
  for (int off = 128; off > 0; off >>= 1) {
    if ((int)threadIdx.x < off) red[threadIdx.x] += red[threadIdx.x + off];
    __syncthreads();
  }
  const float inv = 1.0f / red[0];
  for (int i = threadIdx.x; i < T; i += 256)
    o[i] = f32_to_bf16_rne(__expf(s[i] - mx) * inv);
}

// ---------- layernorm over H, dual f32+bf16 output ----------
__global__ __launch_bounds__(256)
void layernorm_rows_kernel(const float* __restrict__ x, const float* __restrict__ g,
                           const float* __restrict__ be, float* __restrict__ of,
                           unsigned short* __restrict__ ob) {
  const int row = blockIdx.x;
  const float* xr = x + (size_t)row * HDIM;
  __shared__ float rs[256], rq[256];
  float s = 0.0f, q = 0.0f;
  for (int i = threadIdx.x; i < HDIM; i += 256) { float v = xr[i]; s += v; q += v * v; }
  rs[threadIdx.x] = s; rq[threadIdx.x] = q; __syncthreads();
  for (int off = 128; off > 0; off >>= 1) {
    if ((int)threadIdx.x < off) { rs[threadIdx.x] += rs[threadIdx.x + off]; rq[threadIdx.x] += rq[threadIdx.x + off]; }
    __syncthreads();
  }
  const float mu  = rs[0] * (1.0f / HDIM);
  const float var = rq[0] * (1.0f / HDIM) - mu * mu;   // ddof=0 (jnp.var default)
  const float inv = rsqrtf(var + 1e-5f);
  for (int i = threadIdx.x; i < HDIM; i += 256) {
    const float v = (xr[i] - mu) * inv * g[i] + be[i];
    of[(size_t)row * HDIM + i] = v;
    ob[(size_t)row * HDIM + i] = f32_to_bf16_rne(v);
  }
}

// ---------- per-row ||cur - prev||_2 ----------
__global__ __launch_bounds__(256)
void rowdiff_norm_kernel(const float* __restrict__ a, const float* __restrict__ b,
                         float* __restrict__ rown) {
  const int row = blockIdx.x;
  __shared__ float red[256];
  float s = 0.0f;
  for (int i = threadIdx.x; i < HDIM; i += 256) {
    const float d = a[(size_t)row * HDIM + i] - b[(size_t)row * HDIM + i];
    s += d * d;
  }
  red[threadIdx.x] = s; __syncthreads();
  for (int off = 128; off > 0; off >>= 1) {
    if ((int)threadIdx.x < off) red[threadIdx.x] += red[threadIdx.x + off];
    __syncthreads();
  }
  if (threadIdx.x == 0) rown[row] = sqrtf(red[0]);
}

__global__ __launch_bounds__(256)
void conv_reduce_kernel(const float* __restrict__ rown, float* __restrict__ out) {
  __shared__ float red[256];
  float s = 0.0f;
  for (int i = threadIdx.x; i < S_LEN; i += 256) s += rown[i];
  red[threadIdx.x] = s; __syncthreads();
  for (int off = 128; off > 0; off >>= 1) {
    if ((int)threadIdx.x < off) red[threadIdx.x] += red[threadIdx.x + off];
    __syncthreads();
  }
  if (threadIdx.x == 0) {
    const float mean = red[0] * (1.0f / S_LEN);
    out[0] = 1.0f / (1.0f + __expf(10.0f * (mean - 0.01f)));  // sigmoid(-10*(mean-0.01))
  }
}

// ---------- pooled mean over S per column h ----------
__global__ __launch_bounds__(256)
void colmean_kernel(const float* __restrict__ x, float* __restrict__ pooled) {
  const int h = blockIdx.x;
  __shared__ float red[256];
  float s = 0.0f;
  for (int r = threadIdx.x; r < S_LEN; r += 256) s += x[(size_t)r * HDIM + h];
  red[threadIdx.x] = s; __syncthreads();
  for (int off = 128; off > 0; off >>= 1) {
    if ((int)threadIdx.x < off) red[threadIdx.x] += red[threadIdx.x + off];
    __syncthreads();
  }
  if (threadIdx.x == 0) pooled[h] = red[0] * (1.0f / S_LEN);
}

// ---------- global sum / sumsq partials (for unbiased variance) ----------
__global__ __launch_bounds__(256)
void sumsq_partial_kernel(const float* __restrict__ x, long n,
                          float* __restrict__ ps, float* __restrict__ pq) {
  __shared__ float rs[256], rq[256];
  float s = 0.0f, q = 0.0f;
  long i = (long)blockIdx.x * blockDim.x + threadIdx.x;
  const long stride = (long)gridDim.x * blockDim.x;
  for (; i < n; i += stride) { const float v = x[i]; s += v; q += v * v; }
  rs[threadIdx.x] = s; rq[threadIdx.x] = q; __syncthreads();
  for (int off = 128; off > 0; off >>= 1) {
    if ((int)threadIdx.x < off) { rs[threadIdx.x] += rs[threadIdx.x + off]; rq[threadIdx.x] += rq[threadIdx.x + off]; }
    __syncthreads();
  }
  if (threadIdx.x == 0) { ps[blockIdx.x] = rs[0]; pq[blockIdx.x] = rq[0]; }
}

__global__ __launch_bounds__(256)
void finalize_var_kernel(const float* __restrict__ ps, const float* __restrict__ pq,
                         int nb, long n, float* __restrict__ varout) {
  __shared__ float rs[256], rq[256];
  float s = 0.0f, q = 0.0f;
  for (int i = threadIdx.x; i < nb; i += 256) { s += ps[i]; q += pq[i]; }
  rs[threadIdx.x] = s; rq[threadIdx.x] = q; __syncthreads();
  for (int off = 128; off > 0; off >>= 1) {
    if ((int)threadIdx.x < off) { rs[threadIdx.x] += rs[threadIdx.x + off]; rq[threadIdx.x] += rq[threadIdx.x + off]; }
    __syncthreads();
  }
  if (threadIdx.x == 0) {
    const double sum = rs[0], sq = rq[0];
    varout[0] = (float)((sq - sum * sum / (double)n) / (double)(n - 1));  // ddof=1
  }
}

// ---------- tiny complexity-estimator MLP in one block ----------
__global__ __launch_bounds__(512)
void ce_mlp_kernel(const float* __restrict__ pooled,
                   const float* __restrict__ w1, const float* __restrict__ b1,
                   const float* __restrict__ w2, const float* __restrict__ b2,
                   const float* __restrict__ w3, const float* __restrict__ b3,
                   const float* __restrict__ w4, const float* __restrict__ b4,
                   float* __restrict__ complexity) {
  __shared__ float sa[1024];
  __shared__ float sb[512];
  const int t = threadIdx.x;
  for (int i = t; i < 1024; i += 512) sa[i] = pooled[i];
  __syncthreads();
  if (t < 512) { float a = b1[t]; for (int k = 0; k < 1024; ++k) a += w1[t * 1024 + k] * sa[k]; sb[t] = fmaxf(a, 0.0f); }
  __syncthreads();
  if (t < 256) { float a = b2[t]; for (int k = 0; k < 512; ++k) a += w2[t * 512 + k] * sb[k]; sa[t] = fmaxf(a, 0.0f); }
  __syncthreads();
  if (t < 64)  { float a = b3[t]; for (int k = 0; k < 256; ++k) a += w3[t * 256 + k] * sa[k]; sb[t] = fmaxf(a, 0.0f); }
  __syncthreads();
  if (t == 0)  { float a = b4[0]; for (int k = 0; k < 64; ++k) a += w4[k] * sb[k]; complexity[0] = a; }
}

// ---------- scalar heads: required_depth, reasoning_steps, conv_hist[0] ----------
__global__ void finalize_scalars_kernel(const float* __restrict__ complexity,
                                        const float* __restrict__ var,
                                        const float* __restrict__ vd_w,
                                        const float* __restrict__ vd_b,
                                        float* __restrict__ d_out,
                                        long off_rs, long off_conv, long off_rd) {
  if (threadIdx.x == 0 && blockIdx.x == 0) {
    const float total = complexity[0] + logf((float)S_LEN) / 5.0f + (var[0] * vd_w[0] + vd_b[0]);
    const float sig = 1.0f / (1.0f + expf(-total));
    float depth = 1.0f + sig * (float)(DDEPTH - 1);
    depth = fminf(fmaxf(depth, 1.0f), (float)DDEPTH);
    ((int*)d_out)[off_rd] = (int)depth;   // required_depth [1,1] int32
    ((int*)d_out)[off_rs] = 1;            // reasoning_steps [1] int32
    d_out[off_conv] = 0.0f;               // conv_hist[0]
  }
}

// =====================================================================================
extern "C" void kernel_launch(void* const* d_in, const int* in_sizes, int n_in,
                              void* d_out_v, int out_size, void* d_ws, size_t ws_size,
                              hipStream_t stream) {
  (void)in_sizes; (void)n_in; (void)out_size; (void)ws_size;
  const float* hs    = (const float*)d_in[0];
  const float* ce_w1 = (const float*)d_in[1];  const float* ce_b1 = (const float*)d_in[2];
  const float* ce_w2 = (const float*)d_in[3];  const float* ce_b2 = (const float*)d_in[4];
  const float* ce_w3 = (const float*)d_in[5];  const float* ce_b3 = (const float*)d_in[6];
  const float* ce_w4 = (const float*)d_in[7];  const float* ce_b4 = (const float*)d_in[8];
  const float* vd_w  = (const float*)d_in[9];  const float* vd_b  = (const float*)d_in[10];
  const float* rl_w1 = (const float*)d_in[11]; const float* rl_b1 = (const float*)d_in[12];
  const float* rl_w2 = (const float*)d_in[13]; const float* rl_b2 = (const float*)d_in[14];
  const float* rl_g  = (const float*)d_in[15]; const float* rl_be = (const float*)d_in[16];
  const float* mq_w  = (const float*)d_in[17]; const float* mq_b  = (const float*)d_in[18];
  const float* mv_w  = (const float*)d_in[19]; const float* mv_b  = (const float*)d_in[20];
  const float* op_w  = (const float*)d_in[21]; const float* op_b  = (const float*)d_in[22];
  float* d_out = (float*)d_out_v;

  const size_t SM = (size_t)S_LEN * HDIM;            // 2,097,152
  const long n_w1 = (long)DDEPTH * 2 * HDIM * HDIM;  // 25,165,824
  const long n_w2 = n_w1;
  const long n_sq = (long)HDIM * HDIM;

  // d_out layout (flat, return order)
  const long OFF_RS   = (long)SM;              // 2,097,152
  const long OFF_CONV = OFF_RS + 1;
  const long OFF_FM   = OFF_CONV + DDEPTH;     // 2,097,165
  const long OFF_RD   = OFF_FM + 5 * (long)SM; // 12,582,925

  // ---- carve workspace ----
  char* p = (char*)d_ws;
  auto carve = [&](size_t bytes) { char* r = p; p += (bytes + 255) & ~(size_t)255; return r; };
  unsigned short* w1bf   = (unsigned short*)carve((size_t)n_w1 * 2);
  unsigned short* w2bf   = (unsigned short*)carve((size_t)n_w2 * 2);
  unsigned short* mqbf   = (unsigned short*)carve((size_t)n_sq * 2);
  unsigned short* mvbf   = (unsigned short*)carve((size_t)n_sq * 2);
  unsigned short* opbf   = (unsigned short*)carve((size_t)n_sq * 2);
  unsigned short* curbf  = (unsigned short*)carve(SM * 2);
  unsigned short* Z1bf   = (unsigned short*)carve((size_t)S_LEN * 2 * HDIM * 2);
  float*          Zf     = (float*)carve(SM * 4);
  float*          Rf     = (float*)carve(SM * 4);
  unsigned short* Rbf    = (unsigned short*)carve(SM * 2);
  unsigned short* Qbf    = (unsigned short*)carve(SM * 2);
  float*          scores = (float*)carve((size_t)S_LEN * 5 * S_LEN * 4);
  unsigned short* attnbf = (unsigned short*)carve((size_t)S_LEN * 5 * S_LEN * 2);
  float*          curA   = (float*)carve(SM * 4);
  float*          curB   = (float*)carve(SM * 4);
  float*          vtmp   = (float*)carve(SM * 4);
  unsigned short* memPool= (unsigned short*)carve((size_t)DDEPTH * SM * 2);
  float*          rown   = (float*)carve(S_LEN * 4);
  float*          pooled = (float*)carve(HDIM * 4);
  float*          psum   = (float*)carve(512 * 4);
  float*          psq    = (float*)carve(512 * 4);
  float*          cvar   = (float*)carve(4);
  float*          ccomp  = (float*)carve(4);

  // ---- one-time bf16 conversions ----
  cvt_f32_bf16_kernel<<<4096, 256, 0, stream>>>(rl_w1, w1bf, n_w1);
  cvt_f32_bf16_kernel<<<4096, 256, 0, stream>>>(rl_w2, w2bf, n_w2);
  cvt_f32_bf16_kernel<<<1024, 256, 0, stream>>>(mq_w, mqbf, n_sq);
  cvt_f32_bf16_kernel<<<1024, 256, 0, stream>>>(mv_w, mvbf, n_sq);
  cvt_f32_bf16_kernel<<<1024, 256, 0, stream>>>(op_w, opbf, n_sq);
  cvt_f32_bf16_kernel<<<1024, 256, 0, stream>>>(hs, curbf, (long)SM);

  // ---- depth-controller head (tiny) ----
  colmean_kernel<<<HDIM, 256, 0, stream>>>(hs, pooled);
  sumsq_partial_kernel<<<512, 256, 0, stream>>>(hs, (long)SM, psum, psq);
  finalize_var_kernel<<<1, 256, 0, stream>>>(psum, psq, 512, (long)SM, cvar);
  ce_mlp_kernel<<<1, 512, 0, stream>>>(pooled, ce_w1, ce_b1, ce_w2, ce_b2,
                                       ce_w3, ce_b3, ce_w4, ce_b4, ccomp);
  finalize_scalars_kernel<<<1, 32, 0, stream>>>(ccomp, cvar, vd_w, vd_b, d_out,
                                                OFF_RS, OFF_CONV, OFF_RD);

  // ---- reasoning loop ----
  float* cur = curA;
  const dim3 blk(256);
  for (int t = 0; t < DDEPTH; ++t) {
    // z1 = relu(cur_bf @ rl_w1[t]^T + rl_b1[t])   [2048 x 2048], bf16 out
    gemm_bt_wmma<true, true, true><<<dim3(2 * HDIM / 64, S_LEN / 128), blk, 0, stream>>>(
        curbf, HDIM, w1bf + (size_t)t * 2 * HDIM * HDIM, HDIM, 2 * HDIM,
        rl_b1 + (size_t)t * 2 * HDIM, Z1bf, 2 * HDIM, HDIM);
    // z = z1 @ rl_w2[t]^T + rl_b2[t]              [2048 x 1024], f32 out
    gemm_bt_wmma<true, false, false><<<dim3(HDIM / 64, S_LEN / 128), blk, 0, stream>>>(
        Z1bf, 2 * HDIM, w2bf + (size_t)t * HDIM * 2 * HDIM, 2 * HDIM, HDIM,
        rl_b2 + (size_t)t * HDIM, Zf, HDIM, 2 * HDIM);
    // layernorm
    float* ln_f32 = (t == 0) ? cur : Rf;
    layernorm_rows_kernel<<<S_LEN, blk, 0, stream>>>(
        Zf, rl_g + (size_t)t * HDIM, rl_be + (size_t)t * HDIM, ln_f32,
        (t == 0) ? curbf : Rbf);

    float* nxt = cur;
    if (t > 0) {
      nxt = (cur == curA) ? curB : curA;
      const int start = (t - 5 > 0) ? (t - 5) : 0;
      const int T = (t - start) * S_LEN;
      // q = reasoned @ mq^T + mq_b  (bf16)
      gemm_bt_wmma<true, false, true><<<dim3(HDIM / 64, S_LEN / 128), blk, 0, stream>>>(
          Rbf, HDIM, mqbf, HDIM, HDIM, mq_b, Qbf, HDIM, HDIM);
      // scores = q @ memflat^T  (f32)
      gemm_bt_wmma<false, false, false><<<dim3(T / 64, S_LEN / 128), blk, 0, stream>>>(
          Qbf, HDIM, memPool + (size_t)start * SM, HDIM, T, nullptr, scores, T, HDIM);
      softmax_rows_kernel<<<S_LEN, blk, 0, stream>>>(scores, attnbf, T);
      // cur = reasoned + 0.1 * (attn @ memflat)
      gemm_nn_wmma_addscale<<<dim3(HDIM / 64, S_LEN / 128), blk, 0, stream>>>(
          attnbf, T, memPool + (size_t)start * SM, HDIM, T, Rf, nxt, HDIM, T, 0.1f);
      cvt_f32_bf16_kernel<<<1024, 256, 0, stream>>>(nxt, curbf, (long)SM);
      // convergence history
      rowdiff_norm_kernel<<<S_LEN, blk, 0, stream>>>(nxt, cur, rown);
      conv_reduce_kernel<<<1, blk, 0, stream>>>(rown, d_out + OFF_CONV + t);
    }
    // mem value: v = cur_bf @ mv^T + mv_b ; steps >= 7 land in final_memory (f32) directly
    float* vdst = (t >= DDEPTH - 5) ? (d_out + OFF_FM + (size_t)(t - (DDEPTH - 5)) * SM) : vtmp;
    gemm_bt_wmma<true, false, false><<<dim3(HDIM / 64, S_LEN / 128), blk, 0, stream>>>(
        curbf, HDIM, mvbf, HDIM, HDIM, mv_b, vdst, HDIM, HDIM);
    cvt_f32_bf16_kernel<<<1024, 256, 0, stream>>>(vdst, memPool + (size_t)t * SM, (long)SM);
    cur = nxt;
  }

  // output = cur_bf @ op^T + op_b  -> d_out[0 .. S*H)
  gemm_bt_wmma<true, false, false><<<dim3(HDIM / 64, S_LEN / 128), blk, 0, stream>>>(
      curbf, HDIM, opbf, HDIM, HDIM, op_b, d_out, HDIM, HDIM);
}
